// CustomLayerMKM_5385888989194
// MI455X (gfx1250) — matrix operand order repaired and checked by reading, not compiled, not run on hardware
//
#include <hip/hip_runtime.h>
#include <hip/hip_bf16.h>
#include <stdint.h>

// ---------------------------------------------------------------------------
// out = x @ (kron(w0b,w0a)+kron(w1b,w1a)+kron(w2b,w2a))^T + bias
//
// Plan (MI455X reasoning):
//  * Materializing W_sum is ~0.1 GFLOP + 64MB writes; afterwards this is a
//    4096^3 GEMM = 137 GFLOP, far above the 23.3 TB/s roofline (~8 us min
//    traffic) -> compute-bound -> matrix cores.
//  * No fast f32 WMMA on gfx1250 -> bf16x3 split (hi*hi + hi*lo + lo*hi into
//    f32 accumulators) on v_wmma_f32_16x16x32_bf16: fp32-class accuracy at
//    1/3 the bf16 matrix rate.
//  * W kept in natural [o][i] orientation so both LDS tiles are contiguous.
//  * Double-buffered LDS (80 KB of 320 KB/WGP), one barrier per K-step.
//  * NEW: staging uses GLOBAL_LOAD_ASYNC_TO_LDS_B128 (ASYNCcnt path) --
//    memory->LDS DMA with no VGPR round-trip; next tile streams in under the
//    24-WMMA stack, then a single s_wait_asynccnt 0 + barrier.
// ---------------------------------------------------------------------------

typedef __attribute__((ext_vector_type(16))) __bf16 v16bf;
typedef __attribute__((ext_vector_type(8)))  float  v8f;

#define I_DIM 4096
#define O_DIM 4096
#define B_DIM 4096

#define BM 128
#define BN 128
#define BK 32
#define LDSK 40   // padded LDS row stride (bf16): 80B = 5*16B, 16B-aligned rows
#define TILE_ELEMS (BM * LDSK)
#define SMEM_BYTES (4 /*arrays*/ * 2 /*bufs*/ * TILE_ELEMS * 2 /*bytes*/)

union Pack8 { __bf16 h[8]; uint4 q; };
union FragBF { v16bf v; uint4 q[2]; };

// async 16B memory->LDS copy (tracked by ASYNCcnt); LDS address is the low
// 32 bits of the generic pointer (ISA: LDS aperture addr[31:0] = LDS offset)
__device__ __forceinline__ void async_copy_b128(const __bf16* gsrc, __bf16* ldst) {
  unsigned lds_addr = (unsigned)(uintptr_t)ldst;
  asm volatile("global_load_async_to_lds_b128 %0, %1, off"
               :: "v"(lds_addr), "v"(gsrc)
               : "memory");
}
__device__ __forceinline__ void wait_async0() {
  asm volatile("s_wait_asynccnt 0x0" ::: "memory");
}

// ---------------- split x into bf16 hi/lo planes ---------------------------
__global__ void prep_x_kernel(const float* __restrict__ x,
                              __bf16* __restrict__ xhi,
                              __bf16* __restrict__ xlo) {
  const int64_t base = (int64_t)(blockIdx.x * blockDim.x + threadIdx.x) * 8;
  float4 f0 = *(const float4*)(x + base);
  float4 f1 = *(const float4*)(x + base + 4);
  float v[8] = {f0.x, f0.y, f0.z, f0.w, f1.x, f1.y, f1.z, f1.w};
  Pack8 ph, pl;
#pragma unroll
  for (int j = 0; j < 8; ++j) {
    __bf16 h = (__bf16)v[j];
    ph.h[j] = h;
    pl.h[j] = (__bf16)(v[j] - (float)h);
  }
  *(uint4*)(xhi + base) = ph.q;
  *(uint4*)(xlo + base) = pl.q;
}

// ---------------- build W_sum[o][i] (natural orientation), split hi/lo -----
__global__ void prep_w_kernel(const float* __restrict__ w0a, const float* __restrict__ w0b,
                              const float* __restrict__ w1a, const float* __restrict__ w1b,
                              const float* __restrict__ w2a, const float* __restrict__ w2b,
                              __bf16* __restrict__ whi, __bf16* __restrict__ wlo) {
  const int64_t base = (int64_t)(blockIdx.x * blockDim.x + threadIdx.x) * 8;
  const int o  = (int)(base >> 12);   // / 4096
  const int ib = (int)(base & 4095);
  // factor 0: (f1,f2)=(64,64)   -> I1=64,  O1=64
  const int o2_0 = o >> 6, o1_0 = o & 63;
  // factor 1: (f1,f2)=(32,32)   -> I1=128, O1=128
  const int o2_1 = o >> 7, o1_1 = o & 127;
  // factor 2: (f1,f2)=(128,128) -> I1=32,  O1=32
  const int o2_2 = o >> 5, o1_2 = o & 31;
  const float* r0b = w0b + o2_0 * 64;
  const float* r0a = w0a + o1_0 * 64;
  const float* r1b = w1b + o2_1 * 32;
  const float* r1a = w1a + o1_1 * 128;
  const float* r2b = w2b + o2_2 * 128;
  const float* r2a = w2a + o1_2 * 32;
  Pack8 ph, pl;
#pragma unroll
  for (int j = 0; j < 8; ++j) {
    const int i = ib + j;
    const int i2_0 = i >> 6, i1_0 = i & 63;
    const int i2_1 = i >> 7, i1_1 = i & 127;
    const int i2_2 = i >> 5, i1_2 = i & 31;
    float val = r0b[i2_0] * r0a[i1_0]
              + r1b[i2_1] * r1a[i1_1]
              + r2b[i2_2] * r2a[i1_2];
    __bf16 h = (__bf16)val;
    ph.h[j] = h;
    pl.h[j] = (__bf16)(val - (float)h);
  }
  *(uint4*)(whi + base) = ph.q;
  *(uint4*)(wlo + base) = pl.q;
}

// ---------------- bf16x3 GEMM: out[b,o] = bias[o] + sum_i x[b,i]*W[o,i] ----
__global__ __launch_bounds__(256, 1)
void kron_gemm_kernel(const __bf16* __restrict__ xhi, const __bf16* __restrict__ xlo,
                      const __bf16* __restrict__ whi, const __bf16* __restrict__ wlo,
                      const float* __restrict__ bias, float* __restrict__ out) {
  extern __shared__ __bf16 smem[];
  __bf16* sA_hi = smem;                       // [2][BM][LDSK]
  __bf16* sA_lo = sA_hi + 2 * TILE_ELEMS;
  __bf16* sB_hi = sA_lo + 2 * TILE_ELEMS;     // [2][BN][LDSK]  ([n][k])
  __bf16* sB_lo = sB_hi + 2 * TILE_ELEMS;

  const int tid   = threadIdx.x;
  const int lane  = tid & 31;
  const int wave  = tid >> 5;     // 0..7
  const int wm    = wave & 3;     // 4 wave-rows  * 32 m each
  const int wn    = wave >> 2;    // 2 wave-cols  * 64 n each
  const int nlane = lane & 15;
  const int rowBase = blockIdx.y * BM;   // batch rows
  const int colBase = blockIdx.x * BN;   // output cols

  // accumulators preloaded with bias (C element column = nlane)
  v8f acc[2][4];
#pragma unroll
  for (int nt = 0; nt < 4; ++nt) {
    const float bv = bias[colBase + wn * 64 + nt * 16 + nlane];
#pragma unroll
    for (int mt = 0; mt < 2; ++mt) {
#pragma unroll
      for (int r = 0; r < 8; ++r) acc[mt][nt][r] = bv;
    }
  }

  // Per ISA VGPR layouts (CDNA5 ISA §7.12.2):
  // A 16x32 bf16: lanes 0-15 hold K{0..7,16..23}, lanes 16-31 K{8..15,24..31}
  // B 32x16 bf16: lanes 0-15 hold K{0..7,16..23}, lanes 16-31 K{8..15,24..31}
  const int khalf = (lane < 16) ? 0 : 8;
  const int kbase = (lane < 16) ? 0 : 16;

  // stage one BKx(BM|BN) tile pair into LDS buffer `buf` via async DMA
  auto load_tile = [&](int k0, int buf) {
#pragma unroll
    for (int rep = 0; rep < 2; ++rep) {
      const int idx = rep * 256 + tid;
      const int r   = idx >> 2;          // tile row (m or n)
      const int kq  = (idx & 3) * 8;     // k offset within BK
      const int64_t ga = (int64_t)(rowBase + r) * I_DIM + k0 + kq;
      const int64_t gb = (int64_t)(colBase + r) * I_DIM + k0 + kq;
      const int s = buf * TILE_ELEMS + r * LDSK + kq;
      async_copy_b128(xhi + ga, sA_hi + s);
      async_copy_b128(xlo + ga, sA_lo + s);
      async_copy_b128(whi + gb, sB_hi + s);
      async_copy_b128(wlo + gb, sB_lo + s);
    }
  };

  load_tile(0, 0);
  wait_async0();
  __syncthreads();

  for (int k0 = 0; k0 < I_DIM; k0 += BK) {
    const int buf = (k0 / BK) & 1;
    // stream next tile into the other buffer; completes under the WMMA stack
    if (k0 + BK < I_DIM) load_tile(k0 + BK, buf ^ 1);

    const __bf16* Ah = sA_hi + buf * TILE_ELEMS;
    const __bf16* Al = sA_lo + buf * TILE_ELEMS;
    const __bf16* Bh = sB_hi + buf * TILE_ELEMS;
    const __bf16* Bl = sB_lo + buf * TILE_ELEMS;

    FragBF ah[2], al[2];
#pragma unroll
    for (int mt = 0; mt < 2; ++mt) {
      const int m = wm * 32 + mt * 16 + nlane;
      ah[mt].q[0] = *(const uint4*)(Ah + m * LDSK + khalf);
      ah[mt].q[1] = *(const uint4*)(Ah + m * LDSK + 16 + khalf);
      al[mt].q[0] = *(const uint4*)(Al + m * LDSK + khalf);
      al[mt].q[1] = *(const uint4*)(Al + m * LDSK + 16 + khalf);
    }
#pragma unroll
    for (int nt = 0; nt < 4; ++nt) {
      const int n = wn * 64 + nt * 16 + nlane;
      FragBF bh, bl;
      bh.q[0] = *(const uint4*)(Bh + n * LDSK + khalf);
      bh.q[1] = *(const uint4*)(Bh + n * LDSK + 16 + khalf);
      bl.q[0] = *(const uint4*)(Bl + n * LDSK + khalf);
      bl.q[1] = *(const uint4*)(Bl + n * LDSK + 16 + khalf);
#pragma unroll
      for (int mt = 0; mt < 2; ++mt) {
        // bf16x3: hi*hi + hi*lo + lo*hi, f32 accumulate
        acc[mt][nt] = __builtin_amdgcn_wmma_f32_16x16x32_bf16(
            false, ah[mt].v, false, bh.v, (short)0, acc[mt][nt], false, false);
        acc[mt][nt] = __builtin_amdgcn_wmma_f32_16x16x32_bf16(
            false, ah[mt].v, false, bl.v, (short)0, acc[mt][nt], false, false);
        acc[mt][nt] = __builtin_amdgcn_wmma_f32_16x16x32_bf16(
            false, al[mt].v, false, bh.v, (short)0, acc[mt][nt], false, false);
      }
    }
    // own async writes for tile i+1 must land before anyone reads them
    wait_async0();
    __syncthreads();
  }

  // ---- store: C/D layout: VGPR r -> row r (+8 for lanes 16-31), col = nlane
  const int mrow0 = (lane < 16) ? 0 : 8;
#pragma unroll
  for (int mt = 0; mt < 2; ++mt) {
#pragma unroll
    for (int nt = 0; nt < 4; ++nt) {
      const int og = colBase + wn * 64 + nt * 16 + nlane;
#pragma unroll
      for (int r = 0; r < 8; ++r) {
        const int bg = rowBase + wm * 32 + mt * 16 + mrow0 + r;
        out[(int64_t)bg * O_DIM + og] = acc[mt][nt][r];
      }
    }
  }
}

// ---------------------------------------------------------------------------
extern "C" void kernel_launch(void* const* d_in, const int* in_sizes, int n_in,
                              void* d_out, int out_size, void* d_ws, size_t ws_size,
                              hipStream_t stream) {
  const float* x    = (const float*)d_in[0];
  const float* w0a  = (const float*)d_in[1];
  const float* w0b  = (const float*)d_in[2];
  const float* w1a  = (const float*)d_in[3];
  const float* w1b  = (const float*)d_in[4];
  const float* w2a  = (const float*)d_in[5];
  const float* w2b  = (const float*)d_in[6];
  const float* bias = (const float*)d_in[7];
  float* out = (float*)d_out;

  // workspace: 4 bf16 planes of 4096x4096 (32 MB each, 128 MB total)
  __bf16* ws = (__bf16*)d_ws;
  const size_t N = (size_t)B_DIM * I_DIM;
  __bf16* xhi = ws;
  __bf16* xlo = ws + N;
  __bf16* whi = ws + 2 * N;   // W in natural [o][i] layout
  __bf16* wlo = ws + 3 * N;

  const int threads = 256;
  prep_x_kernel<<<dim3((unsigned)(N / (threads * 8))), dim3(threads), 0, stream>>>(x, xhi, xlo);
  prep_w_kernel<<<dim3((unsigned)(((size_t)O_DIM * I_DIM) / (threads * 8))), dim3(threads), 0, stream>>>(
      w0a, w0b, w1a, w1b, w2a, w2b, whi, wlo);
  kron_gemm_kernel<<<dim3(O_DIM / BN, B_DIM / BM), dim3(threads), SMEM_BYTES, stream>>>(
      xhi, xlo, whi, wlo, bias, out);
}
